// FineAggregation_29549374996619
// MI455X (gfx1250) — matrix operand order repaired and checked by reading, not compiled, not run on hardware
//
#include <hip/hip_runtime.h>
#include <math.h>

// ---------------- problem constants ----------------
constexpr int Bc = 2, CF = 32, Hh = 128, Ww = 256;
constexpr int Dd = 5, D7 = 7, CP = 76, CPpad = 96;   // 76 cost planes padded to 3*32
constexpr int HW = Hh * Ww;                           // 32768
constexpr int DHW = Dd * HW;                          // 163840
constexpr int D7HW = D7 * HW;                         // 229376
constexpr int D2 = 3, H2 = 64, W2 = 128;              // stride-2 branch
constexpr int HW2 = H2 * W2, DHW2 = D2 * HW2;         // 8192, 24576
constexpr float EPSf = 1e-5f;

typedef __attribute__((ext_vector_type(16))) _Float16 v16h;
typedef __attribute__((ext_vector_type(8)))  float    v8f;

__device__ __forceinline__ float silu_f(float v) { return v / (1.f + expf(-v)); }

// ---------------------------------------------------------------------------
// 1. disparity samples + warped block-cost volume (76 f16 planes, plane-major)
// ---------------------------------------------------------------------------
__global__ void cost_kernel(const float* __restrict__ left, const float* __restrict__ right,
                            const float* __restrict__ low, const float* __restrict__ high,
                            float* __restrict__ disp_s, _Float16* __restrict__ raw) {
  int idx = blockIdx.x * blockDim.x + threadIdx.x;
  if (idx >= Bc * Dd * HW) return;
  int x = idx % Ww, y = (idx / Ww) % Hh, d = (idx / HW) % Dd, b = idx / (Dd * HW);
  const float sv[5] = {0.f, 3.f / 8.f, 4.f / 8.f, 5.f / 8.f, 1.f};
  size_t pix = (size_t)y * Ww + x;
  float lo = low[(size_t)b * HW + pix], hi = high[(size_t)b * HW + pix];
  float disp = fabsf(hi - lo) * sv[d] + fminf(lo, hi);
  disp_s[((size_t)b * Dd + d) * HW + pix] = disp;

  int x0i[3], x1i[3]; float fr[3], val[3];
  for (int o = 0; o < 3; o++) {
    float coords = (float)x - (disp + (float)(o - 1));
    float x0 = floorf(coords);
    fr[o] = coords - x0;
    float x0c = fminf(fmaxf(x0, 0.f), (float)(Ww - 1));
    float x1c = fminf(fmaxf(x0 + 1.f, 0.f), (float)(Ww - 1));
    x0i[o] = (int)x0c; x1i[o] = (int)x1c;
    val[o] = (coords >= 0.f && coords <= (float)(Ww - 1)) ? 1.f : 0.f;
  }
  float corr[3][4] = {};
  size_t dpix = (size_t)d * HW + pix;
  for (int c = 0; c < CF; c++) {
    float l = left[((size_t)b * CF + c) * HW + pix];
    const float* rrow = right + ((size_t)b * CF + c) * HW + (size_t)y * Ww;
    float wv[3];
    for (int o = 0; o < 3; o++) {
      float g0 = rrow[x0i[o]], g1 = rrow[x1i[o]];
      wv[o] = (g0 * (1.f - fr[o]) + g1 * fr[o]) * val[o];
      corr[o][c >> 3] += l * wv[o];
    }
    raw[((size_t)b * CP + c) * DHW + dpix]      = (_Float16)l;       // lefte
    raw[((size_t)b * CP + 32 + c) * DHW + dpix] = (_Float16)wv[1];   // w0 (offset 0)
  }
  for (int o = 0; o < 3; o++)
    for (int g = 0; g < 4; g++)
      raw[((size_t)b * CP + 64 + o * 4 + g) * DHW + dpix] = (_Float16)(corr[o][g] * 0.125f);
}

// ---------------------------------------------------------------------------
// 2. depthwise 3x3x3 over 76 planes -> channel-last (B, DHW, 96) zero-padded
// ---------------------------------------------------------------------------
__global__ void dwconv76_k(const _Float16* __restrict__ in, const float* __restrict__ w,
                           _Float16* __restrict__ out) {
  size_t idx = (size_t)blockIdx.x * blockDim.x + threadIdx.x;
  if (idx >= (size_t)Bc * DHW * CPpad) return;
  int p  = idx % CPpad;
  int sp = (idx / CPpad) % DHW;
  int b  = idx / ((size_t)CPpad * DHW);
  if (p >= CP) { out[idx] = (_Float16)0.f; return; }   // zero padding channels 76..95
  int x = sp % Ww, y = (sp / Ww) % Hh, z = sp / HW;
  const float* wp = w + p * 27;
  const _Float16* ip = in + ((size_t)b * CP + p) * DHW;
  float acc = 0.f;
  for (int g = 0; g < 27; g++) {
    int kz = g / 9, rr = g % 9, ky = rr / 3, kx = rr % 3;
    int iz = z - 1 + kz, iy = y - 1 + ky, ix = x - 1 + kx;
    if (iz >= 0 && iz < Dd && iy >= 0 && iy < Hh && ix >= 0 && ix < Ww)
      acc += wp[g] * (float)ip[((size_t)iz * Hh + iy) * Ww + ix];
  }
  out[idx] = (_Float16)acc;
}

// ---------------------------------------------------------------------------
// 3. pointwise 76->16 WMMA GEMM; input channel-last (B,DHW,96), unconditional
//    32B v16h B-fragment loads; output channel-last f32 (B,DHW,16)
// ---------------------------------------------------------------------------
__global__ __launch_bounds__(256)
void pw76_wmma_k(const _Float16* __restrict__ in, const float* __restrict__ w,
                 const float* __restrict__ bias, float* __restrict__ out, int tiles) {
  __shared__ _Float16 ldsA[3 * 512];
  int tid = threadIdx.x;
  for (int i = tid; i < 3 * 512; i += 256) {
    int s = i >> 9, r = i & 511, l = r >> 4, e = r & 15;
    int m = l & 15, h = l >> 4;
    int k = h * 16 + e;   // f16 A-fragment K map
    int ci = s * 32 + k;
    ldsA[i] = (_Float16)((ci < CP) ? w[m * CP + ci] : 0.f);
  }
  __syncthreads();
  int wave = tid >> 5, lane = tid & 31;
  int tile = blockIdx.x * 8 + wave;
  if (tile >= tiles) return;
  int b   = tile / (DHW / 16);
  int pix = (tile % (DHW / 16)) * 16 + (lane & 15);
  int h = lane >> 4;
  const _Float16* bp = in + ((size_t)b * DHW + pix) * CPpad;
  v8f acc;
  for (int r = 0; r < 8; r++) acc[r] = bias[8 * h + r];
#pragma unroll
  for (int s = 0; s < 3; s++) {
    v16h A  = *(const v16h*)&ldsA[s * 512 + lane * 16];
    v16h Bf = *(const v16h*)&bp[s * 32 + h * 16];             // one 32B load
    acc = __builtin_amdgcn_wmma_f32_16x16x32_f16(false, A, false, Bf, (short)0, acc, false, false);
  }
  float* op = out + ((size_t)b * DHW + pix) * 16 + 8 * h;
  float4 v0; v0.x = acc[0]; v0.y = acc[1]; v0.z = acc[2]; v0.w = acc[3];
  float4 v1; v1.x = acc[4]; v1.y = acc[5]; v1.z = acc[6]; v1.w = acc[7];
  *(float4*)op = v0; *(float4*)(op + 4) = v1;
}

// ---------------------------------------------------------------------------
// 4. generic 16->16 conv3d implicit-GEMM WMMA, TAPS compile-time (full unroll)
//    input channel-last f16 (B, spatial, 16); output channel-last f32
// ---------------------------------------------------------------------------
template <int TAPS>
__global__ __launch_bounds__(256)
void conv16_wmma_k(const _Float16* __restrict__ in, const float* __restrict__ w,
                   const float* __restrict__ bias, float* __restrict__ out,
                   int Din, int Hin, int Win, int Dout, int Hout, int Wout,
                   int kh, int kw, int stride, int pad, int dil, int tiles) {
  constexpr int STEPS = (TAPS + 1) / 2;             // 2 taps per K=32 WMMA step
  __shared__ _Float16 ldsA[STEPS * 512];
  int tid = threadIdx.x;
  for (int i = tid; i < STEPS * 512; i += 256) {
    int s = i >> 9, r = i & 511, l = r >> 4, e = r & 15;
    int m = l & 15, h = l >> 4;
    int k = h * 16 + e;
    int g = 2 * s + (k >> 4), ci = k & 15;
    ldsA[i] = (_Float16)((g < TAPS) ? w[(m * 16 + ci) * TAPS + g] : 0.f);
  }
  __syncthreads();
  int wave = tid >> 5, lane = tid & 31;
  int tile = blockIdx.x * 8 + wave;
  if (tile >= tiles) return;
  int tw = Wout >> 4;
  int xt = tile % tw, rest = tile / tw;
  int oy = rest % Hout; rest /= Hout;
  int oz = rest % Dout; int b = rest / Dout;
  int n = lane & 15, h = lane >> 4;
  int ox = xt * 16 + n;
  const _Float16* inb = in + (size_t)b * Din * Hin * Win * 16;
  v16h zero;
  for (int e = 0; e < 16; e++) zero[e] = (_Float16)0.f;
  v8f acc;
  for (int r = 0; r < 8; r++) acc[r] = bias ? bias[8 * h + r] : 0.f;
#pragma unroll
  for (int s = 0; s < STEPS; s++) {
    v16h A = *(const v16h*)&ldsA[s * 512 + lane * 16];
    int g = 2 * s + h;                              // tap handled by this lane half
    v16h Bf = zero;
    if (g < TAPS) {
      int kz = g / (kh * kw), rr = g % (kh * kw), ky = rr / kw, kx = rr % kw;
      int iz = oz * stride - pad + kz * dil;
      int iy = oy * stride - pad + ky * dil;
      int ix = ox * stride - pad + kx * dil;
      if (iz >= 0 && iz < Din && iy >= 0 && iy < Hin && ix >= 0 && ix < Win)
        Bf = *(const v16h*)&inb[(((size_t)iz * Hin + iy) * Win + ix) * 16];  // one 32B load
    }
    acc = __builtin_amdgcn_wmma_f32_16x16x32_f16(false, A, false, Bf, (short)0, acc, false, false);
  }
  size_t op = ((size_t)oz * Hout + oy) * Wout + ox;
  float* optr = out + ((size_t)b * Dout * Hout * Wout + op) * 16 + 8 * h;
  float4 v0; v0.x = acc[0]; v0.y = acc[1]; v0.z = acc[2]; v0.w = acc[3];
  float4 v1; v1.x = acc[4]; v1.y = acc[5]; v1.z = acc[6]; v1.w = acc[7];
  *(float4*)optr = v0; *(float4*)(optr + 4) = v1;
}

// ---------------------------------------------------------------------------
// 5. batchnorm over channel-last tensors: deterministic 2-stage reduction
// ---------------------------------------------------------------------------
__global__ void reduce_stats_k(const float* __restrict__ in, float* __restrict__ part, int DHWl) {
  int c = blockIdx.y;
  float s = 0.f, s2 = 0.f;
  int total = Bc * DHWl;
  for (int i = blockIdx.x * blockDim.x + threadIdx.x; i < total; i += gridDim.x * blockDim.x) {
    float v = in[(size_t)i * 16 + c];
    s += v; s2 += v * v;
  }
  __shared__ float sh[256], sh2[256];
  sh[threadIdx.x] = s; sh2[threadIdx.x] = s2; __syncthreads();
  for (int o = 128; o > 0; o >>= 1) {
    if (threadIdx.x < o) { sh[threadIdx.x] += sh[threadIdx.x + o]; sh2[threadIdx.x] += sh2[threadIdx.x + o]; }
    __syncthreads();
  }
  if (threadIdx.x == 0) {
    part[((size_t)c * 64 + blockIdx.x) * 2] = sh[0];
    part[((size_t)c * 64 + blockIdx.x) * 2 + 1] = sh2[0];
  }
}

__global__ void finalize_stats_k(const float* __restrict__ part, float* __restrict__ stats) {
  int c = threadIdx.x;
  if (c >= 16) return;
  float s = 0.f, s2 = 0.f;
  for (int i = 0; i < 64; i++) { s += part[((size_t)c * 64 + i) * 2]; s2 += part[((size_t)c * 64 + i) * 2 + 1]; }
  stats[2 * c] = s; stats[2 * c + 1] = s2;
}

__global__ void bn_act_k(const float* __restrict__ in, const float* __restrict__ stats,
                         const float* __restrict__ gamma, const float* __restrict__ beta,
                         float* __restrict__ out32, _Float16* __restrict__ out16,
                         int DHWl, int doSilu) {
  int idx = blockIdx.x * blockDim.x + threadIdx.x;
  int total = Bc * 16 * DHWl;
  if (idx >= total) return;
  int c = idx & 15;                                   // channel-last
  float N = (float)(Bc * DHWl);
  float m = stats[2 * c] / N;
  float var = stats[2 * c + 1] / N - m * m;
  float v = (in[idx] - m) * rsqrtf(var + EPSf) * gamma[c] + beta[c];
  if (doSilu) v = silu_f(v);
  if (out32) out32[idx] = v;
  if (out16) out16[idx] = (_Float16)v;
}

// ---------------------------------------------------------------------------
// 6. x = silu(x + nearest_up2(y2)) in place (channel-last)
// ---------------------------------------------------------------------------
__global__ void add_up_silu_k(float* __restrict__ x, const float* __restrict__ y2) {
  size_t idx = (size_t)blockIdx.x * blockDim.x + threadIdx.x;
  if (idx >= (size_t)Bc * DHW * 16) return;
  int c  = idx % 16;
  int sp = (idx / 16) % DHW;
  int b  = idx / ((size_t)16 * DHW);
  int xx = sp % Ww, yy = (sp / Ww) % Hh, z = sp / HW;
  size_t sp2 = (size_t)(z >> 1) * HW2 + (size_t)(yy >> 1) * W2 + (xx >> 1);
  float yv = y2[((size_t)b * DHW2 + sp2) * 16 + c];
  float v = x[idx] + yv;
  x[idx] = silu_f(v);
}

// ---------------------------------------------------------------------------
// 7. depthwise 3x3x3 dil=2 pad=2 (f32 channel-last in, f16 channel-last out)
// ---------------------------------------------------------------------------
__global__ void dwconv16d2_k(const float* __restrict__ in, const float* __restrict__ w,
                             _Float16* __restrict__ out) {
  size_t idx = (size_t)blockIdx.x * blockDim.x + threadIdx.x;
  if (idx >= (size_t)Bc * DHW * 16) return;
  int c  = idx % 16;
  int sp = (idx / 16) % DHW;
  int b  = idx / ((size_t)16 * DHW);
  int x = sp % Ww, y = (sp / Ww) % Hh, z = sp / HW;
  const float* ip = in + (size_t)b * DHW * 16;
  const float* wp = w + c * 27;
  float acc = 0.f;
  for (int g = 0; g < 27; g++) {
    int kz = g / 9, rr = g % 9, ky = rr / 3, kx = rr % 3;
    int iz = z - 2 + 2 * kz, iy = y - 2 + 2 * ky, ix = x - 2 + 2 * kx;
    if (iz >= 0 && iz < Dd && iy >= 0 && iy < Hh && ix >= 0 && ix < Ww)
      acc += wp[g] * ip[(((size_t)iz * Hh + iy) * Ww + ix) * 16 + c];
  }
  out[idx] = (_Float16)acc;
}

// ---------------------------------------------------------------------------
// 8. stable 7-way sort + gather cost vectors (32B moves per hypothesis)
// ---------------------------------------------------------------------------
__global__ void cost_sort_k(const float* __restrict__ disp_s, const _Float16* __restrict__ x3,
                            const float* __restrict__ past_b, _Float16* __restrict__ cost,
                            float* __restrict__ ds_out) {
  int idx = blockIdx.x * blockDim.x + threadIdx.x;
  if (idx >= Bc * HW) return;
  int pix = idx % HW, b = idx / HW;
  float v[7]; int id[7];
  for (int d = 0; d < 5; d++) { v[d] = disp_s[((size_t)b * 5 + d) * HW + pix]; id[d] = d; }
  v[5] = 0.f; id[5] = 5; v[6] = 0.f; id[6] = 6;
  for (int i = 1; i < 7; i++) {               // stable insertion sort (ascending)
    float kv = v[i]; int ki = id[i]; int j = i;
    while (j > 0 && v[j - 1] > kv) { v[j] = v[j - 1]; id[j] = id[j - 1]; j--; }
    v[j] = kv; id[j] = ki;
  }
  for (int k = 0; k < 7; k++) ds_out[((size_t)b * 7 + k) * HW + pix] = v[k];
  v16h mvv;
  for (int c = 0; c < 16; c++) mvv[c] = (_Float16)silu_f(past_b[c]);
  for (int k = 0; k < 7; k++) {
    int s = id[k];
    v16h val = (s < 5) ? *(const v16h*)&x3[((size_t)b * DHW + (size_t)s * HW + pix) * 16] : mvv;
    *(v16h*)&cost[((size_t)b * D7HW + (size_t)k * HW + pix) * 16] = val;
  }
}

// ---------------------------------------------------------------------------
// 9. 16->1 conv3d 3x3x3 pad1 (VALU, one v16h per tap), optional tanh (DELTA=1)
// ---------------------------------------------------------------------------
__global__ void conv16to1_k(const _Float16* __restrict__ in, const float* __restrict__ w,
                            float* __restrict__ out, int Dv, int applyTanh) {
  int idx = blockIdx.x * blockDim.x + threadIdx.x;
  if (idx >= Bc * Dv * HW) return;
  int x = idx % Ww, y = (idx / Ww) % Hh, z = (idx / HW) % Dv, b = idx / (Dv * HW);
  const _Float16* ib = in + (size_t)b * Dv * HW * 16;
  float acc = 0.f;
  for (int g = 0; g < 27; g++) {
    int kz = g / 9, rr = g % 9, ky = rr / 3, kx = rr % 3;
    int iz = z - 1 + kz, iy = y - 1 + ky, ix = x - 1 + kx;
    if (iz >= 0 && iz < Dv && iy >= 0 && iy < Hh && ix >= 0 && ix < Ww) {
      v16h hv = *(const v16h*)&ib[(((size_t)iz * Hh + iy) * Ww + ix) * 16];
      for (int ci = 0; ci < 16; ci++) acc += w[ci * 27 + g] * (float)hv[ci];
    }
  }
  out[idx] = applyTanh ? tanhf(acc) : acc;
}

// ---------------------------------------------------------------------------
// 10. top-2 over 7, softmax, expected disparity
// ---------------------------------------------------------------------------
__global__ void topk_k(const float* __restrict__ fc, const float* __restrict__ off,
                       const float* __restrict__ ds, float* __restrict__ dispmap) {
  int idx = blockIdx.x * blockDim.x + threadIdx.x;
  if (idx >= Bc * HW) return;
  int pix = idx % HW, b = idx / HW;
  float f[7];
  for (int k = 0; k < 7; k++) f[k] = fc[((size_t)b * 7 + k) * HW + pix];
  int i0 = 0; float b0 = f[0];
  for (int j = 1; j < 7; j++) if (f[j] > b0) { b0 = f[j]; i0 = j; }
  int i1 = -1; float b1 = -3.4e38f;
  for (int j = 0; j < 7; j++) if (j != i0 && f[j] > b1) { b1 = f[j]; i1 = j; }
  float mx = fmaxf(b0, b1);
  float e0 = expf(b0 - mx), e1 = expf(b1 - mx), inv = 1.f / (e0 + e1);
  float d0 = ds[((size_t)b * 7 + i0) * HW + pix] + off[((size_t)b * 7 + i0) * HW + pix];
  float d1 = ds[((size_t)b * 7 + i1) * HW + pix] + off[((size_t)b * 7 + i1) * HW + pix];
  dispmap[idx] = (e0 * d0 + e1 * d1) * inv;
}

// ---------------------------------------------------------------------------
// 11. mask conv (36ch, 3x3 over left) + softmax-9 + convex 2x upsample
// ---------------------------------------------------------------------------
__global__ void upsample_k(const float* __restrict__ left, const float* __restrict__ up_w,
                           const float* __restrict__ up_b, const float* __restrict__ dispmap,
                           float* __restrict__ out) {
  int idx = blockIdx.x * blockDim.x + threadIdx.x;
  if (idx >= Bc * HW) return;
  int x = idx % Ww, y = (idx / Ww) % Hh, b = idx / HW;
  float m[36];
  for (int co = 0; co < 36; co++) {
    float acc = up_b[co];
    for (int ci = 0; ci < 32; ci++) {
      const float* lp = left + ((size_t)b * 32 + ci) * HW;
      const float* wp = up_w + ((size_t)co * 32 + ci) * 9;
      for (int t = 0; t < 9; t++) {
        int ty = t / 3, tx = t % 3;
        int iy = y - 1 + ty, ix = x - 1 + tx;
        if (iy >= 0 && iy < Hh && ix >= 0 && ix < Ww)
          acc += wp[t] * lp[(size_t)iy * Ww + ix];
      }
    }
    m[co] = acc;
  }
  float patch[9];
  for (int k = 0; k < 9; k++) {
    int ki = k / 3, kj = k % 3;
    int py = y + ki - 1, px = x + kj - 1;
    patch[k] = (py >= 0 && py < Hh && px >= 0 && px < Ww)
                   ? 2.f * dispmap[(size_t)b * HW + (size_t)py * Ww + px] : 0.f;
  }
  for (int i2 = 0; i2 < 2; i2++)
    for (int j2 = 0; j2 < 2; j2++) {
      float mx = -3.4e38f;
      for (int k = 0; k < 9; k++) mx = fmaxf(mx, m[(k * 2 + i2) * 2 + j2]);
      float sum = 0.f, up = 0.f;
      for (int k = 0; k < 9; k++) {
        float e = expf(m[(k * 2 + i2) * 2 + j2] - mx);
        sum += e; up += e * patch[k];
      }
      out[(size_t)b * (4 * HW) + (size_t)(2 * y + i2) * (2 * Ww) + (2 * x + j2)] = up / sum;
    }
}

// ---------------------------------------------------------------------------
// launch
// ---------------------------------------------------------------------------
extern "C" void kernel_launch(void* const* d_in, const int* in_sizes, int n_in,
                              void* d_out, int out_size, void* d_ws, size_t ws_size,
                              hipStream_t stream) {
  (void)in_sizes; (void)n_in; (void)out_size; (void)ws_size;
  const float* left   = (const float*)d_in[0];
  const float* right  = (const float*)d_in[1];
  const float* low    = (const float*)d_in[2];
  const float* high   = (const float*)d_in[3];
  const float* dw1_dw = (const float*)d_in[4];
  const float* dw1_pw = (const float*)d_in[5];
  const float* dw1_b  = (const float*)d_in[6];
  const float* dw1_g  = (const float*)d_in[7];
  const float* dw1_be = (const float*)d_in[8];
  const float* res_w1 = (const float*)d_in[9];
  const float* res_g1 = (const float*)d_in[10];
  const float* res_b1 = (const float*)d_in[11];
  const float* res_w2 = (const float*)d_in[12];
  const float* res_g2 = (const float*)d_in[13];
  const float* res_b2 = (const float*)d_in[14];
  const float* dw2_dw = (const float*)d_in[15];
  const float* dw2_pw = (const float*)d_in[16];
  const float* dw2_g  = (const float*)d_in[17];
  const float* dw2_be = (const float*)d_in[18];
  const float* past_b = (const float*)d_in[21];   // past_w/past_g drop out (conv of zeros)
  const float* cls_w1 = (const float*)d_in[22];
  const float* cls_g1 = (const float*)d_in[23];
  const float* cls_b1 = (const float*)d_in[24];
  const float* cls_w2 = (const float*)d_in[25];
  const float* off_w1 = (const float*)d_in[26];
  const float* off_g1 = (const float*)d_in[27];
  const float* off_b1 = (const float*)d_in[28];
  const float* off_w2 = (const float*)d_in[29];
  const float* up_w   = (const float*)d_in[30];
  const float* up_b   = (const float*)d_in[31];

  float* outp    = (float*)d_out;
  float* up_out  = outp;                       // (B,1,256,512) = 262144
  float* fc_out  = outp + 262144;              // (B,7,H,W)    = 458752
  float* off_out = outp + 262144 + 458752;
  float* ds_out  = outp + 262144 + 2 * 458752;

  // scratch arena (~235 MB)
  size_t o = 0;
  auto arena = [&](size_t bytes) -> char* {
    char* p = (char*)d_ws + o;
    o = (o + bytes + 255) & ~(size_t)255;
    return p;
  };
  float*     stats   = (float*)arena(64 * 4);
  float*     part    = (float*)arena(16 * 64 * 2 * 4);
  float*     disp_s  = (float*)arena((size_t)Bc * 5 * HW * 4);
  float*     dispmap = (float*)arena((size_t)Bc * HW * 4);
  float*     convtmp = (float*)arena((size_t)Bc * D7HW * 16 * 4);
  float*     x32     = (float*)arena((size_t)Bc * DHW * 16 * 4);
  _Float16*  x16     = (_Float16*)arena((size_t)Bc * DHW * 16 * 2);
  _Float16*  y1_16   = (_Float16*)arena((size_t)Bc * DHW2 * 16 * 2);
  float*     y2_32   = (float*)arena((size_t)Bc * DHW2 * 16 * 4);
  _Float16*  dw2d16  = (_Float16*)arena((size_t)Bc * DHW * 16 * 2);
  _Float16*  x3_16   = (_Float16*)arena((size_t)Bc * DHW * 16 * 2);
  _Float16*  cost16  = (_Float16*)arena((size_t)Bc * D7HW * 16 * 2);
  _Float16*  hid16   = (_Float16*)arena((size_t)Bc * D7HW * 16 * 2);
  _Float16*  raw16   = (_Float16*)arena((size_t)Bc * CP * DHW * 2);
  _Float16*  dw1o16  = (_Float16*)arena((size_t)Bc * DHW * CPpad * 2);

  auto runBN = [&](const float* cin, const float* g, const float* be,
                   float* o32, _Float16* o16, int DHWl, int doSilu) {
    dim3 rg(64, 16, 1);
    reduce_stats_k<<<rg, 256, 0, stream>>>(cin, part, DHWl);
    finalize_stats_k<<<1, 16, 0, stream>>>(part, stats);
    int tot = Bc * 16 * DHWl;
    bn_act_k<<<(tot + 255) / 256, 256, 0, stream>>>(cin, stats, g, be, o32, o16, DHWl, doSilu);
  };

  // 1) cost volume
  {
    int tot = Bc * Dd * HW;
    cost_kernel<<<(tot + 255) / 256, 256, 0, stream>>>(left, right, low, high, disp_s, raw16);
  }
  // 2) dw1 depthwise -> channel-last padded (B,DHW,96)
  {
    size_t tot = (size_t)Bc * DHW * CPpad;
    dwconv76_k<<<(unsigned)((tot + 255) / 256), 256, 0, stream>>>(raw16, dw1_dw, dw1o16);
  }
  // 3) dw1 pointwise 76->16 (WMMA) + BN + SiLU
  {
    int tiles = Bc * (DHW / 16);
    pw76_wmma_k<<<(tiles + 7) / 8, 256, 0, stream>>>(dw1o16, dw1_pw, dw1_b, convtmp, tiles);
    runBN(convtmp, dw1_g, dw1_be, x32, x16, DHW, 1);
  }
  // 4) residual branch at half res (WMMA convs)
  {
    int tiles = Bc * D2 * H2 * (W2 / 16);
    conv16_wmma_k<27><<<(tiles + 7) / 8, 256, 0, stream>>>(x16, res_w1, nullptr, convtmp,
                                                           Dd, Hh, Ww, D2, H2, W2, 3, 3, 2, 1, 1, tiles);
    runBN(convtmp, res_g1, res_b1, nullptr, y1_16, DHW2, 1);
    conv16_wmma_k<27><<<(tiles + 7) / 8, 256, 0, stream>>>(y1_16, res_w2, nullptr, convtmp,
                                                           D2, H2, W2, D2, H2, W2, 3, 3, 1, 1, 1, tiles);
    runBN(convtmp, res_g2, res_b2, y2_32, nullptr, DHW2, 0);
  }
  // 5) x = silu(x + up2(y))
  {
    size_t tot = (size_t)Bc * DHW * 16;
    add_up_silu_k<<<(unsigned)((tot + 255) / 256), 256, 0, stream>>>(x32, y2_32);
  }
  // 6) dw2 depthwise dilated + pointwise (WMMA) + BN + SiLU -> x3 (f16)
  {
    size_t tot = (size_t)Bc * DHW * 16;
    dwconv16d2_k<<<(unsigned)((tot + 255) / 256), 256, 0, stream>>>(x32, dw2_dw, dw2d16);
    int tiles = Bc * Dd * Hh * (Ww / 16);
    conv16_wmma_k<1><<<(tiles + 7) / 8, 256, 0, stream>>>(dw2d16, dw2_pw, nullptr, convtmp,
                                                          Dd, Hh, Ww, Dd, Hh, Ww, 1, 1, 1, 0, 1, tiles);
    runBN(convtmp, dw2_g, dw2_be, nullptr, x3_16, DHW, 1);
  }
  // 7) sort hypotheses, assemble 7-deep cost (x3 planes + silu(past_b) memory planes)
  {
    int tot = Bc * HW;
    cost_sort_k<<<(tot + 255) / 256, 256, 0, stream>>>(disp_s, x3_16, past_b, cost16, ds_out);
  }
  // 8) cls head
  {
    int tiles = Bc * D7 * Hh * (Ww / 16);
    conv16_wmma_k<27><<<(tiles + 7) / 8, 256, 0, stream>>>(cost16, cls_w1, nullptr, convtmp,
                                                           D7, Hh, Ww, D7, Hh, Ww, 3, 3, 1, 1, 1, tiles);
    runBN(convtmp, cls_g1, cls_b1, nullptr, hid16, D7HW, 1);
    int tot = Bc * D7 * HW;
    conv16to1_k<<<(tot + 255) / 256, 256, 0, stream>>>(hid16, cls_w2, fc_out, D7, 0);
  }
  // 9) off head (DELTA * tanh, DELTA = 1)
  {
    int tiles = Bc * D7 * Hh * (Ww / 16);
    conv16_wmma_k<27><<<(tiles + 7) / 8, 256, 0, stream>>>(cost16, off_w1, nullptr, convtmp,
                                                           D7, Hh, Ww, D7, Hh, Ww, 3, 3, 1, 1, 1, tiles);
    runBN(convtmp, off_g1, off_b1, nullptr, hid16, D7HW, 1);
    int tot = Bc * D7 * HW;
    conv16to1_k<<<(tot + 255) / 256, 256, 0, stream>>>(hid16, off_w2, off_out, D7, 1);
  }
  // 10) top-2 softmax expectation
  {
    int tot = Bc * HW;
    topk_k<<<(tot + 255) / 256, 256, 0, stream>>>(fc_out, off_out, ds_out, dispmap);
  }
  // 11) convex 2x upsample
  {
    int tot = Bc * HW;
    upsample_k<<<(tot + 255) / 256, 256, 0, stream>>>(left, up_w, up_b, dispmap, up_out);
  }
}
